// IPCA_63410897158337
// MI455X (gfx1250) — matrix operand order repaired and checked by reading, not compiled, not run on hardware
//
#include <hip/hip_runtime.h>

#define B_   4
#define H_   8
#define XL_  1024
#define KL_  256
#define DIM_ 512
#define HD_  64
#define SCALE_ 0.125f

typedef __attribute__((ext_vector_type(16))) __bf16 v16bf;
typedef __attribute__((ext_vector_type(8)))  __bf16 v8bf;
typedef __attribute__((ext_vector_type(8)))  float  v8f;

#if defined(__has_builtin)
#  if __has_builtin(__builtin_amdgcn_global_load_async_to_lds_b128) && \
      __has_builtin(__builtin_amdgcn_s_wait_asynccnt)
#    define USE_ASYNC_LDS 1
#  endif
#endif
#ifndef USE_ASYNC_LDS
#  define USE_ASYNC_LDS 0
#endif

#if USE_ASYNC_LDS
// builtin prototype (from hipcc diagnostic): param 1 is
// 'int __attribute__((vector_size(16))) __device__ *' (global AS), so pass
// exactly that; LDS destination uses AS(3).
typedef int vec4i __attribute__((vector_size(16)));
typedef __attribute__((address_space(1))) vec4i gvec4i;
typedef __attribute__((address_space(3))) vec4i lvec4i;
static __device__ __forceinline__ void async_copy16(const float* g, float* l) {
  // GLOBAL_LOAD_ASYNC_TO_LDS_B128 (ASYNCcnt-tracked DMA, no VGPR round trip)
  __builtin_amdgcn_global_load_async_to_lds_b128((gvec4i*)g, (lvec4i*)l, 0, 0);
}
#endif

static __device__ __forceinline__ v16bf pack16(float4 a, float4 b, float4 c, float4 d) {
  v16bf r;
  r[0]=(__bf16)a.x;  r[1]=(__bf16)a.y;  r[2]=(__bf16)a.z;  r[3]=(__bf16)a.w;
  r[4]=(__bf16)b.x;  r[5]=(__bf16)b.y;  r[6]=(__bf16)b.z;  r[7]=(__bf16)b.w;
  r[8]=(__bf16)c.x;  r[9]=(__bf16)c.y;  r[10]=(__bf16)c.z; r[11]=(__bf16)c.w;
  r[12]=(__bf16)d.x; r[13]=(__bf16)d.y; r[14]=(__bf16)d.z; r[15]=(__bf16)d.w;
  return r;
}

// A fragment (16x32 bf16): lane m = lane&15, K = kk+hsel*16+[0..15]
static __device__ __forceinline__ v16bf load_a_frag(const float* row, int kk, int hsel) {
  float4 f0 = *(const float4*)(row + kk + hsel*16);
  float4 f1 = *(const float4*)(row + kk + hsel*16 + 4);
  float4 f2 = *(const float4*)(row + kk + hsel*16 + 8);
  float4 f3 = *(const float4*)(row + kk + hsel*16 + 12);
  return pack16(f0, f1, f2, f3);
}

// B fragment (32x16 bf16) from 16 contiguous floats: K = (lane>=16?16:0)+e
static __device__ __forceinline__ v16bf load_b_frag_contig(const float* p) {
  float4 f0 = *(const float4*)(p);
  float4 f1 = *(const float4*)(p + 4);
  float4 f2 = *(const float4*)(p + 8);
  float4 f3 = *(const float4*)(p + 12);
  return pack16(f0, f1, f2, f3);
}

static __device__ __forceinline__ v16bf b_frag_from_bf16(const __bf16* row, int off) {
  v8bf lo = *(const v8bf*)(row + off);
  v8bf hi = *(const v8bf*)(row + off + 8);
  v16bf r;
#pragma unroll
  for (int e = 0; e < 8; ++e) { r[e] = lo[e]; r[8+e] = hi[e]; }
  return r;
}

// ---------------------------------------------------------------------------
// Generic bf16-WMMA GEMM: Y[M,N] = (A (+A2)) [M,K] x W[K,N] (+ bias*bias_scale)
// 32x32 output tile / block (4 waves), K stepped by 64 (2 WMMAs per stage).
// Double-buffered LDS; A tiles arrive via async global->LDS DMA (ASYNCcnt)
// when available and no A2-fusion is requested; W tiles are software-
// pipelined through VGPRs (transpose + f32->bf16 on the store).
// a_headed: A stored as [B,H,L,64] pre-merge layout (K = H*64).
// out_mode 0: Out0 row-major [M,N].  out_mode 1: QKV split into Out0/1/2.
// ---------------------------------------------------------------------------
__global__ __launch_bounds__(128) void gemm_bf16_kernel(
    const float* __restrict__ A, const float* __restrict__ A2,
    const float* __restrict__ W, const float* __restrict__ bias, float bias_scale,
    float* __restrict__ Out0, float* __restrict__ Out1, float* __restrict__ Out2,
    int M, int N, int K, int L, int a_headed, int out_mode)
{
  __shared__ float  As[2][32][68];   // f32 A tiles (stride 68 breaks bank conflicts, 16B aligned)
  __shared__ __bf16 Bt[2][32][72];   // transposed bf16 W tiles: Bt[buf][n][k]

  const int tid  = threadIdx.x;
  const int m0   = blockIdx.x * 32;
  const int n0   = blockIdx.y * 32;
  const int lane = tid & 31, wave = tid >> 5;
  const int wm = wave >> 1, wn = wave & 1;
  const int nl = lane & 15, hsel = lane >> 4;
  const int S  = K >> 6;             // number of 64-wide K stages

  // A source address for a 16B-aligned 4-float chunk (headed layout safe:
  // col chunks never cross a 64-wide head block when kk0 is a multiple of 64)
  auto a_src = [&](const float* base, int m, int kg) -> const float* {
    if (a_headed) {
      int bI = m / L, l = m % L;
      int hh = kg >> 6, dd = kg & 63;
      return base + (((size_t)bI * H_ + hh) * (size_t)L + l) * HD_ + dd;
    }
    return base + (size_t)m * K + kg;
  };

  // stage the 32x64 f32 A tile for stage s into buffer buf (512 16B chunks)
  auto stage_A = [&](int s, int buf) {
    const int kk0 = s << 6;
#pragma unroll
    for (int i = 0; i < 4; ++i) {
      int c   = tid + (i << 7);      // 0..511
      int row = c >> 4;
      int col = (c & 15) << 2;
      float* dst = &As[buf][row][col];
      const float* src = a_src(A, m0 + row, kk0 + col);
#if USE_ASYNC_LDS
      if (!A2) { async_copy16(src, dst); continue; }
#endif
      float4 f = *(const float4*)src;
      if (A2) {
        float4 g = *(const float4*)a_src(A2, m0 + row, kk0 + col);
        f.x += g.x; f.y += g.y; f.z += g.z; f.w += g.w;
      }
      *(float4*)dst = f;
    }
  };

  // W tile pipeline: 64 k-rows x 32 n-cols, each thread owns 16 floats
  const int wrow  = tid >> 1;          // 0..63
  const int wcol0 = (tid & 1) << 4;    // 0 or 16
  float4 wcur[4], wnxt[4];

  {
    const float* src = W + (size_t)wrow * N + n0 + wcol0;
#pragma unroll
    for (int i = 0; i < 4; ++i) wcur[i] = *(const float4*)(src + 4*i);
  }
  stage_A(0, 0);

  v8f acc = {};

  for (int s = 0; s < S; ++s) {
    const int buf = s & 1;
    // issue next stage (buffers freed by the post-compute barrier of s-1)
    if (s + 1 < S) {
      stage_A(s + 1, buf ^ 1);
      const float* src = W + (size_t)((s + 1) * 64 + wrow) * N + n0 + wcol0;
#pragma unroll
      for (int i = 0; i < 4; ++i) wnxt[i] = *(const float4*)(src + 4*i);
    }
    // store current W tile transposed as bf16
#pragma unroll
    for (int i = 0; i < 4; ++i) {
      Bt[buf][wcol0 + 4*i + 0][wrow] = (__bf16)wcur[i].x;
      Bt[buf][wcol0 + 4*i + 1][wrow] = (__bf16)wcur[i].y;
      Bt[buf][wcol0 + 4*i + 2][wrow] = (__bf16)wcur[i].z;
      Bt[buf][wcol0 + 4*i + 3][wrow] = (__bf16)wcur[i].w;
    }
#if USE_ASYNC_LDS
    if (!A2) {
      if (s + 1 < S) __builtin_amdgcn_s_wait_asynccnt(4);  // stage s DMA done
      else           __builtin_amdgcn_s_wait_asynccnt(0);
    }
#endif
    __syncthreads();   // staged data visible to all waves

    const float*  arow = &As[buf][wm*16 + nl][0];
    const __bf16* brow = &Bt[buf][wn*16 + nl][0];
#pragma unroll
    for (int kkk = 0; kkk < 64; kkk += 32) {
      v16bf a  = load_a_frag(arow, kkk, hsel);
      v16bf bf = b_frag_from_bf16(brow, kkk + hsel*16);
      acc = __builtin_amdgcn_wmma_f32_16x16x32_bf16(false, a, false, bf,
                                                    (short)0, acc, false, false);
    }
    __syncthreads();   // frees buffer `buf` for stage s+2
#pragma unroll
    for (int i = 0; i < 4; ++i) wcur[i] = wnxt[i];
  }

  // ---- store D (lane n = nl; VGPR v -> row v + 8*hsel) ----
  const int n_g = n0 + wn*16 + nl;
  const float bv = bias ? bias[n_g] * bias_scale : 0.0f;
#pragma unroll
  for (int v = 0; v < 8; ++v) {
    int m_g = m0 + wm*16 + v + 8*hsel;
    float val = acc[v] + bv;
    if (out_mode == 0) {
      Out0[(size_t)m_g * N + n_g] = val;
    } else {
      int which = n_g >> 9;          // /512 -> q,k,v
      int hh    = (n_g >> 6) & 7;    // head
      int dd    = n_g & 63;
      float* o  = (which == 0) ? Out0 : ((which == 1) ? Out1 : Out2);
      int bI = m_g / L, l = m_g % L;
      o[(((size_t)bI * H_ + hh) * (size_t)L + l) * HD_ + dd] = val;
    }
  }
}

// ---------------------------------------------------------------------------
// Fused attention, rows = KL (k-direction softmax over x).
// S = (Qh Kh^T)*scale -> (optional) polar-bin |S| sums -> main_ori -> mask/bias
// -> softmax over XL -> O = P Vh.   16 rows per block, 4 waves, S in LDS.
// ---------------------------------------------------------------------------
__global__ __launch_bounds__(128) void attn_rows_kl(
    const float* __restrict__ Qh,   // [B,H,KL,64]
    const float* __restrict__ Kh,   // [B,H,XL,64]
    const float* __restrict__ Vh,   // [B,H,XL,64]
    const unsigned char* __restrict__ mask, // [B,H,XL,KL], used transposed
    float maskval,
    const int* __restrict__ rd, const int* __restrict__ polar_pos,
    const float* __restrict__ dis_embed, const float* __restrict__ polar_emb,
    int* __restrict__ main_ori_out,
    float* __restrict__ Opre,       // [B,H,KL,64]
    int use_bias)
{
  __shared__ float S[16][XL_];     // 64 KB score strip

  const int bh = blockIdx.y;
  const int b  = bh >> 3, h = bh & 7;
  const int r0 = blockIdx.x * 16;
  const int tid  = threadIdx.x;
  const int lane = tid & 31, wave = tid >> 5;
  const int nl = lane & 15, hsel = lane >> 4;

  // hoisted Q fragments (shared by every column tile of this wave)
  const float* qrow = Qh + ((size_t)bh * KL_ + (r0 + nl)) * HD_;
  v16bf qa0 = load_a_frag(qrow, 0,  hsel);
  v16bf qa1 = load_a_frag(qrow, 32, hsel);

  // ---- phase 1: score strip ----
  for (int ct = wave; ct < XL_/16; ct += 4) {
    int c0 = ct * 16;
    const float* kp = Kh + ((size_t)bh * XL_ + (c0 + nl)) * HD_;
    v16bf b0 = load_b_frag_contig(kp + hsel*16);
    v16bf b1 = load_b_frag_contig(kp + 32 + hsel*16);
    v8f s = {};
    s = __builtin_amdgcn_wmma_f32_16x16x32_bf16(false, qa0, false, b0, (short)0, s, false, false);
    s = __builtin_amdgcn_wmma_f32_16x16x32_bf16(false, qa1, false, b1, (short)0, s, false, false);
#pragma unroll
    for (int v = 0; v < 8; ++v) S[v + 8*hsel][c0 + nl] = s[v] * SCALE_;
  }
  __syncthreads();

  // ---- phase 2: per-row ori / bias / mask / softmax (one wave per row) ----
  for (int rr = wave; rr < 16; rr += 4) {
    const int row = r0 + rr;
    int mo = 0;
    if (use_bias) {
      float bins[8] = {0.f,0.f,0.f,0.f,0.f,0.f,0.f,0.f};
      for (int c = lane; c < XL_; c += 32) {
        float sv = S[rr][c];
        int p = polar_pos[((size_t)b * KL_ + row) * XL_ + c];
        p = p < 0 ? 0 : (p > 7 ? 7 : p);
        bins[p] += fabsf(sv);
      }
      for (int o = 16; o; o >>= 1)
#pragma unroll
        for (int j = 0; j < 8; ++j) bins[j] += __shfl_xor(bins[j], o, 32);
      float best = bins[0]; mo = 0;
#pragma unroll
      for (int j = 1; j < 8; ++j) if (bins[j] > best) { best = bins[j]; mo = j; }
      if (lane == 0 && main_ori_out) main_ori_out[(size_t)bh * KL_ + row] = mo;
    }
    float lmax = -3.0e38f;
    for (int c = lane; c < XL_; c += 32) {
      float sv = S[rr][c];
      if (mask[((size_t)bh * XL_ + c) * KL_ + row]) sv = maskval;
      if (use_bias) {
        size_t pidx = ((size_t)b * KL_ + row) * XL_ + c;
        int rv = rd[pidx];
        int p  = polar_pos[pidx];
        p = p < 0 ? 0 : (p > 7 ? 7 : p);
        sv += dis_embed[rv * H_ + h] + polar_emb[(p - mo + 8) & 7];
      }
      S[rr][c] = sv;
      lmax = fmaxf(lmax, sv);
    }
    for (int o = 16; o; o >>= 1) lmax = fmaxf(lmax, __shfl_xor(lmax, o, 32));
    float lsum = 0.f;
    for (int c = lane; c < XL_; c += 32) {
      float e = __expf(S[rr][c] - lmax);
      S[rr][c] = e; lsum += e;
    }
    for (int o = 16; o; o >>= 1) lsum += __shfl_xor(lsum, o, 32);
    float inv = 1.0f / lsum;
    for (int c = lane; c < XL_; c += 32) S[rr][c] *= inv;
  }
  __syncthreads();

  // ---- phase 3: O = P @ V  (wave -> 16-wide d tile) ----
  const int d0 = wave * 16;
  v8f acc = {};
  for (int kb = 0; kb < XL_; kb += 32) {
    v16bf a = load_a_frag(&S[nl][0], kb, hsel);
    v16bf bfrag;
    const float* vp = Vh + ((size_t)bh * XL_ + kb + hsel*16) * HD_ + d0 + nl;
#pragma unroll
    for (int e = 0; e < 16; ++e) bfrag[e] = (__bf16)vp[(size_t)e * HD_];
    acc = __builtin_amdgcn_wmma_f32_16x16x32_bf16(false, a, false, bfrag,
                                                  (short)0, acc, false, false);
  }
#pragma unroll
  for (int v = 0; v < 8; ++v)
    Opre[((size_t)bh * KL_ + r0 + v + 8*hsel) * HD_ + d0 + nl] = acc[v];
}

// ---------------------------------------------------------------------------
// Fused attention, rows = XL (x-direction softmax over k). mask direct,
// optional transposed bias reading precomputed main_ori.
// ---------------------------------------------------------------------------
__global__ __launch_bounds__(128) void attn_rows_xl(
    const float* __restrict__ Qh,   // [B,H,XL,64]
    const float* __restrict__ Kh,   // [B,H,KL,64]
    const float* __restrict__ Vh,   // [B,H,KL,64]
    const unsigned char* __restrict__ mask, // [B,H,XL,KL]
    const int* __restrict__ rd, const int* __restrict__ polar_pos,
    const float* __restrict__ dis_embed, const float* __restrict__ polar_emb,
    const int* __restrict__ main_ori_in,
    float* __restrict__ Opre,       // [B,H,XL,64]
    int use_bias)
{
  __shared__ float S[16][KL_];     // 16 KB

  const int bh = blockIdx.y;
  const int b  = bh >> 3, h = bh & 7;
  const int r0 = blockIdx.x * 16;
  const int tid  = threadIdx.x;
  const int lane = tid & 31, wave = tid >> 5;
  const int nl = lane & 15, hsel = lane >> 4;

  const float* qrow = Qh + ((size_t)bh * XL_ + (r0 + nl)) * HD_;
  v16bf qa0 = load_a_frag(qrow, 0,  hsel);
  v16bf qa1 = load_a_frag(qrow, 32, hsel);

  for (int ct = wave; ct < KL_/16; ct += 4) {
    int c0 = ct * 16;
    const float* kp = Kh + ((size_t)bh * KL_ + (c0 + nl)) * HD_;
    v16bf b0 = load_b_frag_contig(kp + hsel*16);
    v16bf b1 = load_b_frag_contig(kp + 32 + hsel*16);
    v8f s = {};
    s = __builtin_amdgcn_wmma_f32_16x16x32_bf16(false, qa0, false, b0, (short)0, s, false, false);
    s = __builtin_amdgcn_wmma_f32_16x16x32_bf16(false, qa1, false, b1, (short)0, s, false, false);
#pragma unroll
    for (int v = 0; v < 8; ++v) S[v + 8*hsel][c0 + nl] = s[v] * SCALE_;
  }
  __syncthreads();

  for (int rr = wave; rr < 16; rr += 4) {
    const int row = r0 + rr;
    float lmax = -3.0e38f;
    for (int c = lane; c < KL_; c += 32) {
      float sv = S[rr][c];
      if (mask[((size_t)bh * XL_ + row) * KL_ + c]) sv = -1.0e9f;
      if (use_bias) {
        size_t pidx = ((size_t)b * KL_ + c) * XL_ + row;
        int rv = rd[pidx];
        int p  = polar_pos[pidx];
        p = p < 0 ? 0 : (p > 7 ? 7 : p);
        int mo = main_ori_in[(size_t)bh * KL_ + c];
        sv += dis_embed[rv * H_ + h] + polar_emb[(p - mo + 8) & 7];
      }
      S[rr][c] = sv;
      lmax = fmaxf(lmax, sv);
    }
    for (int o = 16; o; o >>= 1) lmax = fmaxf(lmax, __shfl_xor(lmax, o, 32));
    float lsum = 0.f;
    for (int c = lane; c < KL_; c += 32) {
      float e = __expf(S[rr][c] - lmax);
      S[rr][c] = e; lsum += e;
    }
    for (int o = 16; o; o >>= 1) lsum += __shfl_xor(lsum, o, 32);
    float inv = 1.0f / lsum;
    for (int c = lane; c < KL_; c += 32) S[rr][c] *= inv;
  }
  __syncthreads();

  const int d0 = wave * 16;
  v8f acc = {};
  for (int kb = 0; kb < KL_; kb += 32) {
    v16bf a = load_a_frag(&S[nl][0], kb, hsel);
    v16bf bfrag;
    const float* vp = Vh + ((size_t)bh * KL_ + kb + hsel*16) * HD_ + d0 + nl;
#pragma unroll
    for (int e = 0; e < 16; ++e) bfrag[e] = (__bf16)vp[(size_t)e * HD_];
    acc = __builtin_amdgcn_wmma_f32_16x16x32_bf16(false, a, false, bfrag,
                                                  (short)0, acc, false, false);
  }
#pragma unroll
  for (int v = 0; v < 8; ++v)
    Opre[((size_t)bh * XL_ + r0 + v + 8*hsel) * HD_ + d0 + nl] = acc[v];
}

// ---------------------------------------------------------------------------
extern "C" void kernel_launch(void* const* d_in, const int* in_sizes, int n_in,
                              void* d_out, int out_size, void* d_ws, size_t ws_size,
                              hipStream_t stream) {
  const float* x        = (const float*)d_in[0];
  const float* kernal   = (const float*)d_in[1];
  const float* i_x      = (const float*)d_in[2];
  const float* i_kernal = (const float*)d_in[3];
  const int*   rd        = (const int*)d_in[4];
  const int*   polar_pos = (const int*)d_in[5];
  const unsigned char* att_mask       = (const unsigned char*)d_in[6];
  const unsigned char* i_att_mask     = (const unsigned char*)d_in[7];
  const unsigned char* cross_att_mask = (const unsigned char*)d_in[8];
  const float* Wqkv     = (const float*)d_in[9];
  const float* Wqkv_i   = (const float*)d_in[10];
  const float* Wqkv_i2  = (const float*)d_in[11];
  const float* Wproj    = (const float*)d_in[12];
  const float* bproj    = (const float*)d_in[13];
  const float* Wproj_i  = (const float*)d_in[14];
  const float* bproj_i  = (const float*)d_in[15];
  const float* Wproj_i2 = (const float*)d_in[16];
  const float* bproj_i2 = (const float*)d_in[17];
  const float* dis_embed = (const float*)d_in[18];
  const float* polar_emb = (const float*)d_in[19];

  float* x_out   = (float*)d_out;                               // [B,XL,512]
  float* k_out   = x_out + (size_t)B_ * XL_ * DIM_;             // [B,KL,512]
  float* i_x_out = k_out + (size_t)B_ * KL_ * DIM_;             // [B,XL,512]

  char* ws = (char*)d_ws;
  auto alloc = [&](size_t nbytes) -> float* {
    float* p = (float*)ws;
    ws += ((nbytes + 255) / 256) * 256;
    return p;
  };
  const size_t XLH = (size_t)B_ * H_ * XL_ * HD_ * sizeof(float);
  const size_t KLH = (size_t)B_ * H_ * KL_ * HD_ * sizeof(float);

  float* qh_x  = alloc(XLH); float* kh_x  = alloc(XLH); float* vh_x  = alloc(XLH);
  float* qh_kn = alloc(KLH); float* kh_kn = alloc(KLH); float* vh_kn = alloc(KLH);
  float* qh_ix = alloc(XLH); float* kh_ix = alloc(XLH); float* vh_ix = alloc(XLH);
  float* qh_ikn = alloc(KLH); float* kh_ikn = alloc(KLH); float* vh_ikn = alloc(KLH);
  float* qh_ik2 = alloc(KLH); float* kh_ik2 = alloc(KLH); float* vh_ik2 = alloc(KLH);
  float* pre_kout  = alloc(KLH);
  float* pre_o1    = alloc(XLH);
  float* pre_o2    = alloc(XLH);
  float* pre_ikout = alloc(KLH);
  float* ikout_m   = alloc((size_t)B_ * KL_ * DIM_ * sizeof(float));
  float* pre_ixo   = alloc(XLH);
  int*   main_ori  = (int*)alloc((size_t)B_ * H_ * KL_ * sizeof(int));

  const dim3 blk(128);

  // --- QKV projections (bf16 WMMA GEMMs with fused split/transpose) ---
  gemm_bf16_kernel<<<dim3((B_*XL_)/32, (3*DIM_)/32), blk, 0, stream>>>(
      x, nullptr, Wqkv, nullptr, 0.f, qh_x, kh_x, vh_x,
      B_*XL_, 3*DIM_, DIM_, XL_, 0, 1);
  gemm_bf16_kernel<<<dim3((B_*KL_)/32, (3*DIM_)/32), blk, 0, stream>>>(
      kernal, nullptr, Wqkv, nullptr, 0.f, qh_kn, kh_kn, vh_kn,
      B_*KL_, 3*DIM_, DIM_, KL_, 0, 1);
  gemm_bf16_kernel<<<dim3((B_*XL_)/32, (3*DIM_)/32), blk, 0, stream>>>(
      i_x, nullptr, Wqkv_i, nullptr, 0.f, qh_ix, kh_ix, vh_ix,
      B_*XL_, 3*DIM_, DIM_, XL_, 0, 1);
  gemm_bf16_kernel<<<dim3((B_*KL_)/32, (3*DIM_)/32), blk, 0, stream>>>(
      i_kernal, nullptr, Wqkv_i, nullptr, 0.f, qh_ikn, kh_ikn, vh_ikn,
      B_*KL_, 3*DIM_, DIM_, KL_, 0, 1);

  // --- kx attention (computes main_ori) and ik_ix attention ---
  attn_rows_kl<<<dim3(KL_/16, B_*H_), blk, 0, stream>>>(
      qh_kn, kh_x, vh_x, att_mask, -1.0e6f, rd, polar_pos, dis_embed, polar_emb,
      main_ori, pre_kout, 1);
  attn_rows_kl<<<dim3(KL_/16, B_*H_), blk, 0, stream>>>(
      qh_ikn, kh_ix, vh_ix, i_att_mask, -1.0e9f, nullptr, nullptr, dis_embed,
      polar_emb, nullptr, pre_ikout, 0);

  // --- x_k (transposed bias via main_ori) and x_ik attentions ---
  attn_rows_xl<<<dim3(XL_/16, B_*H_), blk, 0, stream>>>(
      qh_x, kh_kn, vh_kn, att_mask, rd, polar_pos, dis_embed, polar_emb,
      main_ori, pre_o1, 1);
  attn_rows_xl<<<dim3(XL_/16, B_*H_), blk, 0, stream>>>(
      qh_x, kh_ikn, vh_ikn, cross_att_mask, nullptr, nullptr, dis_embed,
      polar_emb, nullptr, pre_o2, 0);

  // --- output projections (merge fused into A addressing) ---
  gemm_bf16_kernel<<<dim3((B_*KL_)/32, DIM_/32), blk, 0, stream>>>(
      pre_kout, nullptr, Wproj, bproj, 1.f, k_out, nullptr, nullptr,
      B_*KL_, DIM_, DIM_, KL_, 1, 0);
  gemm_bf16_kernel<<<dim3((B_*XL_)/32, DIM_/32), blk, 0, stream>>>(
      pre_o1, pre_o2, Wproj, bproj, 2.f, x_out, nullptr, nullptr,
      B_*XL_, DIM_, DIM_, XL_, 1, 0);
  gemm_bf16_kernel<<<dim3((B_*KL_)/32, DIM_/32), blk, 0, stream>>>(
      pre_ikout, nullptr, Wproj_i, bproj_i, 1.f, ikout_m, nullptr, nullptr,
      B_*KL_, DIM_, DIM_, KL_, 1, 0);

  // --- second-stage: qkv(i_k_out), ix_ik2 attention, final projection ---
  gemm_bf16_kernel<<<dim3((B_*KL_)/32, (3*DIM_)/32), blk, 0, stream>>>(
      ikout_m, nullptr, Wqkv_i2, nullptr, 0.f, qh_ik2, kh_ik2, vh_ik2,
      B_*KL_, 3*DIM_, DIM_, KL_, 0, 1);
  attn_rows_xl<<<dim3(XL_/16, B_*H_), blk, 0, stream>>>(
      qh_ix, kh_ik2, vh_ik2, i_att_mask, nullptr, nullptr, dis_embed,
      polar_emb, nullptr, pre_ixo, 0);
  gemm_bf16_kernel<<<dim3((B_*XL_)/32, DIM_/32), blk, 0, stream>>>(
      pre_ixo, nullptr, Wproj_i2, bproj_i2, 1.f, i_x_out, nullptr, nullptr,
      B_*XL_, DIM_, DIM_, XL_, 1, 0);
}
